// MemoryNet_52553219833997
// MI455X (gfx1250) — matrix operand order repaired and checked by reading, not compiled, not run on hardware
//
#include <hip/hip_runtime.h>

typedef __attribute__((ext_vector_type(16))) __bf16 v16bf;
typedef __attribute__((ext_vector_type(8)))  __bf16 v8bf;
typedef __attribute__((ext_vector_type(4)))  __bf16 v4bf;
typedef __attribute__((ext_vector_type(8)))  float  v8f;

#define HD    128        // head dim (1024 / 8 heads)
#define LSEQ  2048       // sequence length
#define KC    64         // key-chunk (2x WMMA K-dim)
#define QW    16         // queries per wave (WMMA M)
#define WAVES 4

// LDS strides in bf16 elements; +8 elems (16B) pad spreads banks, keeps 16B align
#define KSTR  136        // K tile row: l-major, 128 d + pad   (64 rows)
#define VSTR  72         // V tile row: d-major, 64 l + pad    (128 rows)
#define QSTR  136        // Q tile row: q-major, 128 d + pad   (16 rows/wave)
#define PSTR  72         // P scratch row: q-major, 64 l + pad (16 rows/wave)

// cross-lane xor reduction within 16-lane halves via ds_swizzle (imm xor mask)
__device__ __forceinline__ float swz_max(float x) {
    float y;
    y = __int_as_float(__builtin_amdgcn_ds_swizzle(__float_as_int(x), 0x041F));
    x = fmaxf(x, y);
    y = __int_as_float(__builtin_amdgcn_ds_swizzle(__float_as_int(x), 0x081F));
    x = fmaxf(x, y);
    y = __int_as_float(__builtin_amdgcn_ds_swizzle(__float_as_int(x), 0x101F));
    x = fmaxf(x, y);
    y = __int_as_float(__builtin_amdgcn_ds_swizzle(__float_as_int(x), 0x201F));
    return fmaxf(x, y);
}
__device__ __forceinline__ float swz_sum(float x) {
    x += __int_as_float(__builtin_amdgcn_ds_swizzle(__float_as_int(x), 0x041F));
    x += __int_as_float(__builtin_amdgcn_ds_swizzle(__float_as_int(x), 0x081F));
    x += __int_as_float(__builtin_amdgcn_ds_swizzle(__float_as_int(x), 0x101F));
    x += __int_as_float(__builtin_amdgcn_ds_swizzle(__float_as_int(x), 0x201F));
    return x;
}

// A-layout fragment: 16 contiguous bf16/lane at {p..p+15}
__device__ __forceinline__ v16bf ldfragA(const __bf16* p) {
    v16bf r;
    *(v8bf*)&r       = *(const v8bf*)(p);
    *((v8bf*)&r + 1) = *(const v8bf*)(p + 8);
    return r;
}
// B-layout fragment: 16 contiguous bf16/lane at {p..p+15}
__device__ __forceinline__ v16bf ldfragB(const __bf16* p) {
    v16bf r;
    *(v8bf*)&r       = *(const v8bf*)(p);
    *((v8bf*)&r + 1) = *(const v8bf*)(p + 8);
    return r;
}

__device__ __forceinline__ v4bf pack4(float a, float b, float c, float d) {
    v4bf r;
    r[0] = (__bf16)a; r[1] = (__bf16)b; r[2] = (__bf16)c; r[3] = (__bf16)d;
    return r;
}

__global__ __launch_bounds__(128) void
memnet_attn_fwd(const float* __restrict__ qg, const float* __restrict__ kg,
                const float* __restrict__ vg, const float* __restrict__ gamma,
                float* __restrict__ outg)
{
    __shared__ __attribute__((aligned(16))) __bf16 klds[KC * KSTR];
    __shared__ __attribute__((aligned(16))) __bf16 vlds[HD * VSTR];
    __shared__ __attribute__((aligned(16))) __bf16 qlds[WAVES][QW * QSTR];
    __shared__ __attribute__((aligned(16))) __bf16 plds[WAVES][QW * PSTR];

    const int tid  = threadIdx.x;
    const int wave = tid >> 5;
    const int lane = tid & 31;
    const int half = lane >> 4;      // which 16-lane half
    const int ln   = lane & 15;      // lane within half

    const int qblk = blockIdx.x & 31;    // L / 64 query blocks
    const int bh   = blockIdx.x >> 5;    // 0..31 (b*8 + h)
    const int q0   = qblk * (QW * WAVES) + wave * QW;

    const float scale = 0.08838834764831845f; // 1/sqrt(128)

    const float* qp = qg + (size_t)bh * HD * LSEQ;
    const float* kp = kg + (size_t)bh * HD * LSEQ;
    const float* vp = vg + (size_t)bh * HD * LSEQ;
    float*       op = outg + (size_t)bh * HD * LSEQ;

    // ---- stage this wave's 16x128 Q tile (pre-scaled bf16), 4 d packed / store
    {
        __bf16* qb = qlds[wave] + ln * QSTR;
        const float* qsrc = qp + q0 + ln;
        for (int d4 = half * 4; d4 < HD; d4 += 8) {
            const float* p = qsrc + (size_t)d4 * LSEQ;
            v4bf b = pack4(p[0] * scale, p[LSEQ] * scale,
                           p[2 * LSEQ] * scale, p[3 * LSEQ] * scale);
            *(v4bf*)(qb + d4) = b;
        }
    }

    // ---- 4 persistent Q A-fragments (16x32, K = d-chunk); wave-private LDS,
    // DS ops are in-order within a wave so no barrier needed.
    v16bf qa[4];
    {
        const __bf16* qb = qlds[wave] + ln * QSTR + half * 16;
#pragma unroll
        for (int c = 0; c < 4; ++c) qa[c] = ldfragA(qb + c * 32);
    }

    // ---- running state: rows M = r + 8*half (replicated over 16 lanes)
    float m[8], s[8];
    v8f   o[8];
    const v8f vz = {0.f, 0.f, 0.f, 0.f, 0.f, 0.f, 0.f, 0.f};
#pragma unroll
    for (int r = 0; r < 8; ++r) { m[r] = -1e30f; s[r] = 0.f; }
#pragma unroll
    for (int t = 0; t < 8; ++t) o[t] = vz;

    const int kl   = tid & 63;          // K fill: l position (0..63)
    const int kd0  = (tid >> 6) * 4;    // K fill: starting d4 (0 or 4)
    const int vl4  = (tid & 15) * 4;    // V fill: l4 position (0..60)
    const int vd0  = tid >> 4;          // V fill: starting d row (0..7)

    for (int l0 = 0; l0 < LSEQ; l0 += KC) {
        // ---- K tile (l-major): 4 strided-d loads -> one packed ds_store_b64
        {
            const float* ks = kp + l0 + kl;
            __bf16*      kd = klds + kl * KSTR;
            for (int d4 = kd0; d4 < HD; d4 += 8) {
                const float* p = ks + (size_t)d4 * LSEQ;
                *(v4bf*)(kd + d4) =
                    pack4(p[0], p[LSEQ], p[2 * LSEQ], p[3 * LSEQ]);
            }
        }
        // ---- V tile (d-major): float4 load -> one packed ds_store_b64
        {
            const float* vs = vp + l0 + vl4;
            __bf16*      vd = vlds + vl4;
            for (int d = vd0; d < HD; d += 8) {
                float4 f = *(const float4*)(vs + (size_t)d * LSEQ);
                *(v4bf*)(vd + d * VSTR) = pack4(f.x, f.y, f.z, f.w);
            }
        }
        __syncthreads();

        // ---- S = Q * K^T : four 16x16 f32 tiles; batch 8 ds_loads per 4 WMMAs
        v8f st[4];
#pragma unroll
        for (int t = 0; t < 4; ++t) {
            const __bf16* kb = klds + (t * 16 + ln) * KSTR + half * 16;
            v16bf f0 = ldfragB(kb);
            v16bf f1 = ldfragB(kb + 32);
            v16bf f2 = ldfragB(kb + 64);
            v16bf f3 = ldfragB(kb + 96);
            v8f acc = vz;
            acc = __builtin_amdgcn_wmma_f32_16x16x32_bf16(false, qa[0], false, f0, (short)0, acc, false, false);
            acc = __builtin_amdgcn_wmma_f32_16x16x32_bf16(false, qa[1], false, f1, (short)0, acc, false, false);
            acc = __builtin_amdgcn_wmma_f32_16x16x32_bf16(false, qa[2], false, f2, (short)0, acc, false, false);
            acc = __builtin_amdgcn_wmma_f32_16x16x32_bf16(false, qa[3], false, f3, (short)0, acc, false, false);
            st[t] = acc;
        }

        // ---- online softmax over 64 keys at once
        float corr[8];
#pragma unroll
        for (int r = 0; r < 8; ++r) {
            float x = fmaxf(fmaxf(st[0][r], st[1][r]), fmaxf(st[2][r], st[3][r]));
            x = swz_max(x);
            const float mn = fmaxf(m[r], x);
            corr[r] = __expf(m[r] - mn);
            m[r]    = mn;
        }
#pragma unroll
        for (int r = 0; r < 8; ++r) {
            const float p0 = __expf(st[0][r] - m[r]);
            const float p1 = __expf(st[1][r] - m[r]);
            const float p2 = __expf(st[2][r] - m[r]);
            const float p3 = __expf(st[3][r] - m[r]);
            st[0][r] = p0; st[1][r] = p1; st[2][r] = p2; st[3][r] = p3;
            s[r] = s[r] * corr[r] + swz_sum((p0 + p1) + (p2 + p3));
        }

        // ---- C-layout P -> LDS (bf16)
        {
            __bf16* pb = plds[wave] + (half * 8) * PSTR + ln;
#pragma unroll
            for (int r = 0; r < 8; ++r) {
                pb[r * PSTR]      = (__bf16)st[0][r];
                pb[r * PSTR + 16] = (__bf16)st[1][r];
                pb[r * PSTR + 32] = (__bf16)st[2][r];
                pb[r * PSTR + 48] = (__bf16)st[3][r];
            }
        }

        // ---- rescale running O while P stores are in flight
#pragma unroll
        for (int t = 0; t < 8; ++t)
#pragma unroll
            for (int r = 0; r < 8; ++r)
                o[t][r] *= corr[r];

        // ---- reload P as two 16x32 A-fragments (l-chunks)
        v16bf pa[2];
        {
            const __bf16* pb = plds[wave] + ln * PSTR + half * 16;
            pa[0] = ldfragA(pb);
            pa[1] = ldfragA(pb + 32);
        }

        // ---- O += P * V : 8 d-tiles x 2 l-chunks; batch 8 ds_loads per 4 WMMAs
#pragma unroll
        for (int t = 0; t < 8; t += 2) {
            const __bf16* vb0 = vlds + (t * 16 + ln) * VSTR + half * 16;
            const __bf16* vb1 = vb0 + 16 * VSTR;
            v16bf f00 = ldfragB(vb0);
            v16bf f01 = ldfragB(vb0 + 32);
            v16bf f10 = ldfragB(vb1);
            v16bf f11 = ldfragB(vb1 + 32);
            o[t]     = __builtin_amdgcn_wmma_f32_16x16x32_bf16(false, pa[0], false, f00, (short)0, o[t],     false, false);
            o[t]     = __builtin_amdgcn_wmma_f32_16x16x32_bf16(false, pa[1], false, f01, (short)0, o[t],     false, false);
            o[t + 1] = __builtin_amdgcn_wmma_f32_16x16x32_bf16(false, pa[0], false, f10, (short)0, o[t + 1], false, false);
            o[t + 1] = __builtin_amdgcn_wmma_f32_16x16x32_bf16(false, pa[1], false, f11, (short)0, o[t + 1], false, false);
        }
        __syncthreads();
    }

    // ---- epilogue: scale by gamma / rowsum, store (d, q) as 32B segments
    const float g = *gamma;
    float fin[8];
#pragma unroll
    for (int r = 0; r < 8; ++r) fin[r] = g / s[r];

#pragma unroll
    for (int t = 0; t < 8; ++t) {
        float* dst = op + (size_t)(t * 16 + ln) * LSEQ + q0 + half * 8;
        float4 a = make_float4(o[t][0] * fin[0], o[t][1] * fin[1],
                               o[t][2] * fin[2], o[t][3] * fin[3]);
        float4 b = make_float4(o[t][4] * fin[4], o[t][5] * fin[5],
                               o[t][6] * fin[6], o[t][7] * fin[7]);
        *(float4*)dst       = a;
        *((float4*)dst + 1) = b;
    }
}

extern "C" void kernel_launch(void* const* d_in, const int* in_sizes, int n_in,
                              void* d_out, int out_size, void* d_ws, size_t ws_size,
                              hipStream_t stream) {
    const float* q     = (const float*)d_in[0];
    const float* k     = (const float*)d_in[1];
    const float* v     = (const float*)d_in[2];
    const float* gamma = (const float*)d_in[3];
    float* out = (float*)d_out;

    // grid: (B*H = 32) x (L / 64-query block = 32) = 1024 blocks, 128 thr (4 waves)
    dim3 grid(32 * 32);
    memnet_attn_fwd<<<grid, 128, 0, stream>>>(q, k, v, gamma, out);
}
